// SSAN_34875134443868
// MI455X (gfx1250) — matrix operand order repaired and checked by reading, not compiled, not run on hardware
//
#include <hip/hip_runtime.h>
#include <math.h>

// ---------------------------------------------------------------------------
// SSAN masked-attention pipeline for MI455X (gfx1250, wave32, WMMA).
// fp32 operands pre-split into bf16 hi/lo arrays (same bytes as fp32) so the
// WMMA GEMM hot loops are pure global_load_b128 + v_wmma_f32_16x16x32_bf16.
// Wave tile 32x64 (MT=2 x NT=4). Per k-step all operands (A hi/lo x2,
// B hi/lo x4) are loaded first, then 24 WMMAs issue in three passes of 8 so
// each accumulator is revisited only after 8 independent WMMAs (>= the 5-slot
// bf16 WMMA->WMMA hazard window) -> no v_nop stalls. 32-bit offsets keep
// addressing in GVS (sgpr base + 32-bit voffset + immediate) form.
// ---------------------------------------------------------------------------

typedef __attribute__((ext_vector_type(16))) __bf16 v16bf;
typedef __attribute__((ext_vector_type(8)))  __bf16 v8bf;
typedef __attribute__((ext_vector_type(4)))  __bf16 v4bf;
typedef __attribute__((ext_vector_type(8)))  float  v8f;

#define N_ROWS 8192
#define D_DIM  256
#define PE_W   0.8f
#define SCALE  0.0625f    // 1/sqrt(256)
#define TOPN   1024

__device__ __forceinline__ v16bf cat8(v8bf a, v8bf b)
{
    return __builtin_shufflevector(a, b, 0, 1, 2, 3, 4, 5, 6, 7,
                                          8, 9, 10, 11, 12, 13, 14, 15);
}

// ---------------------------------------------------------------------------
// Elementwise split: v = c0*s0[i] + c1*s1[i];  hi = bf16(v); lo = bf16(v-hi)
// ---------------------------------------------------------------------------
__global__ __launch_bounds__(256)
void SSAN_split_kernel(const float* __restrict__ s0, const float* __restrict__ s1,
                       float c0, float c1,
                       __bf16* __restrict__ hi, __bf16* __restrict__ lo, int n4)
{
    const int i = blockIdx.x * 256 + threadIdx.x;
    if (i >= n4) return;
    const float4 a = ((const float4*)s0)[i];
    const float4 b = ((const float4*)s1)[i];
    float v[4];
    v[0] = c0 * a.x + c1 * b.x;
    v[1] = c0 * a.y + c1 * b.y;
    v[2] = c0 * a.z + c1 * b.z;
    v[3] = c0 * a.w + c1 * b.w;
    v4bf vh, vl;
#pragma unroll
    for (int e = 0; e < 4; ++e) {
        const __bf16 h = (__bf16)v[e];
        vh[e] = h;
        vl[e] = (__bf16)(v[e] - (float)h);
    }
    ((v4bf*)hi)[i] = vh;
    ((v4bf*)lo)[i] = vl;
}

// ---------------------------------------------------------------------------
// Per-wave NT-GEMM on pre-split bf16 operands:
//   C[i,j] = sum_d A[i,d]*B[j,d], wave tile 16*MT(M) x 16*NT(N), fp32 acc.
// VGPR layouts used here (wave32, k contiguous per lane half):
//   A (16x32): lane m+16h, elems 0..15 -> K=d0+16h+e (row i0+m, two 16B chunks)
//   B (32x16): lane n+16h, elems 0..15 -> K=d0+16h+e (row j0+n, contiguous 32B)
//   C (16x16): vgpr c, lane n+16h -> row i0+c+8h, col j0+n
// ---------------------------------------------------------------------------
template<int MT, int NT, int DK>
__device__ __forceinline__ void wave_gemm_bf16(
    const __bf16* __restrict__ Ahi, const __bf16* __restrict__ Alo,
    const __bf16* __restrict__ Bhi, const __bf16* __restrict__ Blo,
    int lda, int ldb, int i0, int j0, v8f acc[MT][NT])
{
    const int lane = threadIdx.x & 31;
    const int m    = lane & 15;
    const int h    = lane >> 4;

    unsigned arow[MT];
#pragma unroll
    for (int mt = 0; mt < MT; ++mt)
        arow[mt] = (unsigned)(i0 + 16 * mt + m) * (unsigned)lda;
    unsigned brow[NT];
#pragma unroll
    for (int t = 0; t < NT; ++t)
        brow[t] = (unsigned)(j0 + 16 * t + m) * (unsigned)ldb;

#pragma unroll
    for (int d0 = 0; d0 < DK; d0 += 32) {
        const unsigned oa0 = d0 + 16 * h;       // 16B aligned chunk
        const unsigned oa1 = d0 + 16 * h + 8;   // 16B aligned chunk
        const unsigned ob  = d0 + 16 * h;       // 32B aligned chunk

        // ---- load everything for this k-step ------------------------
        v16bf ahi[MT], alo[MT];
#pragma unroll
        for (int mt = 0; mt < MT; ++mt) {
            ahi[mt] = cat8(*(const v8bf*)(Ahi + arow[mt] + oa0),
                           *(const v8bf*)(Ahi + arow[mt] + oa1));
            alo[mt] = cat8(*(const v8bf*)(Alo + arow[mt] + oa0),
                           *(const v8bf*)(Alo + arow[mt] + oa1));
        }
        v16bf bhi[NT], blo[NT];
#pragma unroll
        for (int t = 0; t < NT; ++t) {
            bhi[t] = *(const v16bf*)(Bhi + brow[t] + ob);
            blo[t] = *(const v16bf*)(Blo + brow[t] + ob);
        }

        // ---- 3 passes x MT*NT accumulators: same-acc WMMAs are
        //      MT*NT (=8) slots apart -> no WMMA->WMMA hazard NOPs ----
#pragma unroll
        for (int t = 0; t < NT; ++t)
#pragma unroll
            for (int mt = 0; mt < MT; ++mt)
                acc[mt][t] = __builtin_amdgcn_wmma_f32_16x16x32_bf16(
                    false, alo[mt], false, bhi[t], (short)0, acc[mt][t], false, false);
#pragma unroll
        for (int t = 0; t < NT; ++t)
#pragma unroll
            for (int mt = 0; mt < MT; ++mt)
                acc[mt][t] = __builtin_amdgcn_wmma_f32_16x16x32_bf16(
                    false, ahi[mt], false, blo[t], (short)0, acc[mt][t], false, false);
#pragma unroll
        for (int t = 0; t < NT; ++t)
#pragma unroll
            for (int mt = 0; mt < MT; ++mt)
                acc[mt][t] = __builtin_amdgcn_wmma_f32_16x16x32_bf16(
                    false, ahi[mt], false, bhi[t], (short)0, acc[mt][t], false, false);
    }
}

template<int MT, int NT>
__device__ __forceinline__ void zero_acc(v8f acc[MT][NT])
{
    for (int mt = 0; mt < MT; ++mt)
        for (int t = 0; t < NT; ++t)
#pragma unroll
            for (int e = 0; e < 8; ++e) acc[mt][t][e] = 0.0f;
}

// ---------------------------------------------------------------------------
// Kernel: P[i,j] = sum_d r[i,d]*W[j,d] + r[i,j]  (r = ca*x + cb*pe, fp32),
// writing P directly as bf16 hi/lo for the downstream att GEMM.
// Grid (D/64, N/128), block 128 (4 waves), wave tile 32x64.
// ---------------------------------------------------------------------------
__global__ __launch_bounds__(128)
void SSAN_proj_kernel(const __bf16* __restrict__ Rhi, const __bf16* __restrict__ Rlo,
                      const __bf16* __restrict__ Whi, const __bf16* __restrict__ Wlo,
                      const float* __restrict__ x, const float* __restrict__ pe,
                      float ca, float cb,
                      __bf16* __restrict__ Phi, __bf16* __restrict__ Plo)
{
    const int wave = threadIdx.x >> 5;
    const int lane = threadIdx.x & 31;
    const int i0 = blockIdx.y * 128 + wave * 32;
    const int j0 = blockIdx.x * 64;

    v8f acc[2][4];
    zero_acc<2, 4>(acc);
    wave_gemm_bf16<2, 4, D_DIM>(Rhi, Rlo, Whi, Wlo, D_DIM, D_DIM, i0, j0, acc);

    const int h = lane >> 4;
    const int n = lane & 15;
#pragma unroll
    for (int mt = 0; mt < 2; ++mt)
#pragma unroll
        for (int t = 0; t < 4; ++t)
#pragma unroll
            for (int c = 0; c < 8; ++c) {
                const int row = i0 + 16 * mt + c + 8 * h;
                const int col = j0 + 16 * t + n;
                const unsigned idx = (unsigned)row * D_DIM + col;
                const float q = acc[mt][t][c] + ca * x[idx] + cb * pe[idx];
                const __bf16 hv = (__bf16)q;
                Phi[idx] = hv;
                Plo[idx] = (__bf16)(q - (float)hv);
            }
}

// ---------------------------------------------------------------------------
// Kernel: PS[i,j] = (pe_q[i,:] . pe_kv[j,:]) * scale  (fp32 out, 256MB)
// ---------------------------------------------------------------------------
__global__ __launch_bounds__(128)
void SSAN_pesims_kernel(const __bf16* __restrict__ PQhi, const __bf16* __restrict__ PQlo,
                        const __bf16* __restrict__ PKhi, const __bf16* __restrict__ PKlo,
                        float* __restrict__ ps)
{
    const int wave = threadIdx.x >> 5;
    const int lane = threadIdx.x & 31;
    const int i0 = blockIdx.y * 128 + wave * 32;
    const int j0 = blockIdx.x * 64;

    v8f acc[2][4];
    zero_acc<2, 4>(acc);
    wave_gemm_bf16<2, 4, D_DIM>(PQhi, PQlo, PKhi, PKlo, D_DIM, D_DIM, i0, j0, acc);

    const int h = lane >> 4;
    const int n = lane & 15;
#pragma unroll
    for (int mt = 0; mt < 2; ++mt)
#pragma unroll
        for (int t = 0; t < 4; ++t)
#pragma unroll
            for (int c = 0; c < 8; ++c) {
                const int row = i0 + 16 * mt + c + 8 * h;
                const int col = j0 + 16 * t + n;
                const unsigned idx = (unsigned)row * N_ROWS + col;
                ps[idx] = acc[mt][t][c] * SCALE;
            }
}

// ---------------------------------------------------------------------------
// Kernel: per-row 1024-th largest of PS row, 4-pass radix select.
// Row staged once in LDS (32KB) as order-preserving u32 keys.
// ---------------------------------------------------------------------------
__global__ __launch_bounds__(256)
void SSAN_thresh_kernel(const float* __restrict__ ps, float* __restrict__ thr)
{
    __shared__ unsigned int keys[N_ROWS];
    __shared__ unsigned int hist[256];
    __shared__ unsigned int s_prefix, s_k;

    const int row = blockIdx.x;
    const int tid = threadIdx.x;

    for (int t = tid; t < N_ROWS; t += 256) {
        unsigned int u = __float_as_uint(ps[(size_t)row * N_ROWS + t]);
        u = (u & 0x80000000u) ? ~u : (u | 0x80000000u);   // total order
        keys[t] = u;
    }
    if (tid == 0) { s_prefix = 0u; s_k = TOPN; }
    __syncthreads();

    for (int pass = 3; pass >= 0; --pass) {
        const int shift = pass * 8;
        hist[tid] = 0u;
        __syncthreads();
        const unsigned int prefix = s_prefix;
        const unsigned int pmask  = (pass == 3) ? 0u : (0xFFFFFFFFu << (shift + 8));
        for (int t = tid; t < N_ROWS; t += 256) {
            const unsigned int u = keys[t];
            if ((u & pmask) == (prefix & pmask))
                atomicAdd(&hist[(u >> shift) & 0xFFu], 1u);
        }
        __syncthreads();
        if (tid == 0) {
            unsigned int k = s_k, cum = 0u;
            int sel = 0;
            for (int b = 255; b >= 0; --b) {
                if (cum + hist[b] >= k) { sel = b; k -= cum; break; }
                cum += hist[b];
            }
            s_k = k;
            s_prefix = prefix | ((unsigned int)sel << shift);
        }
        __syncthreads();
    }

    if (tid == 0) {
        const unsigned int u = s_prefix;
        const unsigned int b = (u & 0x80000000u) ? (u & 0x7FFFFFFFu) : ~u;
        thr[row] = __uint_as_float(b);
    }
}

// ---------------------------------------------------------------------------
// Kernel: att[i,j] = (PS[i,j] < thr[i]) ? 0 : (Q[i,:].K[j,:]) * scale
// Output stored non-temporally (write-once, keep L2 for resident operands).
// ---------------------------------------------------------------------------
__global__ __launch_bounds__(128)
void SSAN_att_kernel(const __bf16* __restrict__ Qhi, const __bf16* __restrict__ Qlo,
                     const __bf16* __restrict__ Khi, const __bf16* __restrict__ Klo,
                     const float* __restrict__ ps, const float* __restrict__ thr,
                     float* __restrict__ out)
{
    const int wave = threadIdx.x >> 5;
    const int lane = threadIdx.x & 31;
    const int i0 = blockIdx.y * 128 + wave * 32;
    const int j0 = blockIdx.x * 64;

    v8f acc[2][4];
    zero_acc<2, 4>(acc);
    wave_gemm_bf16<2, 4, D_DIM>(Qhi, Qlo, Khi, Klo, D_DIM, D_DIM, i0, j0, acc);

    const int h = lane >> 4;
    const int n = lane & 15;
#pragma unroll
    for (int mt = 0; mt < 2; ++mt)
#pragma unroll
        for (int t = 0; t < 4; ++t)
#pragma unroll
            for (int c = 0; c < 8; ++c) {
                const int row = i0 + 16 * mt + c + 8 * h;
                const int col = j0 + 16 * t + n;
                const unsigned idx = (unsigned)row * N_ROWS + col;
                const float s = ps[idx];
                const float v = (s < thr[row]) ? 0.0f : acc[mt][t][c] * SCALE;
                __builtin_nontemporal_store(v, out + idx);
            }
}

// ---------------------------------------------------------------------------
// Launcher.  Workspace layout:
//   PS   fp32 [8192 x 8192]                              256 MB
//   12 x bf16 [8192 x 256] hi/lo arrays                   48 MB
//     (PQ, PK, RQ, RK, Q, K) x (hi, lo)
//   4 x bf16 [256 x 256]  W splits                        0.5 MB
//   thr  fp32 [8192]                                      32 KB
// ---------------------------------------------------------------------------
extern "C" void kernel_launch(void* const* d_in, const int* in_sizes, int n_in,
                              void* d_out, int out_size, void* d_ws, size_t ws_size,
                              hipStream_t stream)
{
    const float* x_q   = (const float*)d_in[0];
    const float* x_kv  = (const float*)d_in[1];
    const float* pe_q  = (const float*)d_in[2];
    const float* pe_kv = (const float*)d_in[3];
    const float* W_q   = (const float*)d_in[4];
    const float* W_k   = (const float*)d_in[5];
    // d_in[6] = W_v, d_in[7] = b_v: value path unused by the reference output.
    float* out = (float*)d_out;

    char* ws = (char*)d_ws;
    const size_t ps_bytes = (size_t)N_ROWS * N_ROWS * sizeof(float);  // 256 MB
    const size_t nd_bytes = (size_t)N_ROWS * D_DIM * sizeof(__bf16);  // 4 MB
    const size_t w_bytes  = (size_t)D_DIM * D_DIM * sizeof(__bf16);   // 128 KB

    float* PS = (float*)ws;
    char* p = ws + ps_bytes;
    __bf16* PQhi = (__bf16*)(p + 0 * nd_bytes);
    __bf16* PQlo = (__bf16*)(p + 1 * nd_bytes);
    __bf16* PKhi = (__bf16*)(p + 2 * nd_bytes);
    __bf16* PKlo = (__bf16*)(p + 3 * nd_bytes);
    __bf16* RQhi = (__bf16*)(p + 4 * nd_bytes);
    __bf16* RQlo = (__bf16*)(p + 5 * nd_bytes);
    __bf16* RKhi = (__bf16*)(p + 6 * nd_bytes);
    __bf16* RKlo = (__bf16*)(p + 7 * nd_bytes);
    __bf16* Qhi  = (__bf16*)(p + 8 * nd_bytes);
    __bf16* Qlo  = (__bf16*)(p + 9 * nd_bytes);
    __bf16* Khi  = (__bf16*)(p + 10 * nd_bytes);
    __bf16* Klo  = (__bf16*)(p + 11 * nd_bytes);
    char* pw = p + 12 * nd_bytes;
    __bf16* WQhi = (__bf16*)(pw + 0 * w_bytes);
    __bf16* WQlo = (__bf16*)(pw + 1 * w_bytes);
    __bf16* WKhi = (__bf16*)(pw + 2 * w_bytes);
    __bf16* WKlo = (__bf16*)(pw + 3 * w_bytes);
    float* thr = (float*)(pw + 4 * w_bytes);

    const float ca = fabsf(1.0f - PE_W);  // |1-w| in fp32, matches reference
    const float cb = PE_W;

    // ---- operand splits (elementwise, L2-resident afterwards) -----------
    const int nd4 = N_ROWS * D_DIM / 4;   // 524288
    const int w4  = D_DIM * D_DIM / 4;    // 16384
    const dim3 sblk(256);
    SSAN_split_kernel<<<dim3(nd4 / 256), sblk, 0, stream>>>(pe_q,  pe_q,  1.f, 0.f, PQhi, PQlo, nd4);
    SSAN_split_kernel<<<dim3(nd4 / 256), sblk, 0, stream>>>(pe_kv, pe_kv, 1.f, 0.f, PKhi, PKlo, nd4);
    SSAN_split_kernel<<<dim3(nd4 / 256), sblk, 0, stream>>>(x_q,  pe_q,  ca, cb, RQhi, RQlo, nd4);
    SSAN_split_kernel<<<dim3(nd4 / 256), sblk, 0, stream>>>(x_kv, pe_kv, ca, cb, RKhi, RKlo, nd4);
    SSAN_split_kernel<<<dim3(w4 / 256),  sblk, 0, stream>>>(W_q, W_q, 1.f, 0.f, WQhi, WQlo, w4);
    SSAN_split_kernel<<<dim3(w4 / 256),  sblk, 0, stream>>>(W_k, W_k, 1.f, 0.f, WKhi, WKlo, w4);

    // ---- WMMA GEMMs + select + mask -------------------------------------
    const dim3 blk(128);
    const dim3 gProj(D_DIM / 64, N_ROWS / 128);    // (4, 64)
    const dim3 gBig(N_ROWS / 64, N_ROWS / 128);    // (128, 64)

    SSAN_proj_kernel<<<gProj, blk, 0, stream>>>(RQhi, RQlo, WQhi, WQlo, x_q,  pe_q,  ca, cb, Qhi, Qlo);
    SSAN_proj_kernel<<<gProj, blk, 0, stream>>>(RKhi, RKlo, WKhi, WKlo, x_kv, pe_kv, ca, cb, Khi, Klo);
    SSAN_pesims_kernel<<<gBig, blk, 0, stream>>>(PQhi, PQlo, PKhi, PKlo, PS);
    SSAN_thresh_kernel<<<dim3(N_ROWS), dim3(256), 0, stream>>>(PS, thr);
    SSAN_att_kernel<<<gBig, blk, 0, stream>>>(Qhi, Qlo, Khi, Klo, PS, thr, out);
}
